// DeepMDsimpleForces_32495722561836
// MI455X (gfx1250) — compile-verified, matrix operand order checked by reading, not run on hardware
//
#include <hip/hip_runtime.h>
#include <hip/hip_bf16.h>
#include <math.h>

// ---------------- static model config ----------------
#define NSAMP 32
#define NPP   10                 // Np
#define NCELLS 50
#define NPTS  (NCELLS*NPP)       // 500
#define NATOM (NSAMP*NPTS)       // 16000
#define NFK   1001
#define LBOX  50.0f
#define KSTEP (6.28318530717958647692f / LBOX)
#define LRAV0 0.26040068f
#define LRAV1 0.01500177f
#define LRSTD0 0.0147799f
#define LRSTD1 0.00377196f
// AV = (0,0), STD = (1,1) kept literal below.

typedef __attribute__((ext_vector_type(16))) _Float16 v16h;
typedef __attribute__((ext_vector_type(8)))  _Float16 v8h;
typedef __attribute__((ext_vector_type(8)))  float    v8f;
typedef __attribute__((ext_vector_type(4)))  int      v4i;

struct Net5  { const float* W[5]; const float* b[5]; };
struct FitNet{ const float* W[4]; const float* b[4]; const float* lw; };

// async global->LDS path (gfx1250): guarded so compile never regresses
#if defined(__AMDGCN__) && __has_builtin(__builtin_amdgcn_global_load_async_to_lds_b128) && __has_builtin(__builtin_amdgcn_s_wait_asynccnt)
#define USE_ASYNC_LDS 1
#else
#define USE_ASYNC_LDS 0
#endif

// ---------------- workspace layout (float offsets) ----------------
#define WS_FRE  0
#define WS_FIM  (WS_FRE + NSAMP*NFK)
#define WS_URE  (WS_FIM + NSAMP*NFK)
#define WS_UIM  (WS_URE + NATOM)
#define WS_GD   (WS_UIM + NATOM)        // NATOM*64 f32
#define WS_GL3  (WS_GD  + NATOM*64)     // NATOM*32 f32
#define WS_GUR  (WS_GL3 + NATOM*32)
#define WS_GUI  (WS_GUR + NATOM)
#define WS_CC   (WS_GUI + NATOM)
#define WS_CS   (WS_CC  + NSAMP*NFK)
#define WS_MKA  (WS_CS  + NSAMP*NFK)
#define WS_MKB  (WS_MKA + NSAMP*NFK)
#define WS_F16  (((WS_MKB + NSAMP*NFK) + 7) & ~7)   // 32B-aligned f16 region
// f16 region: X0ws (NATOM*96 halves) then wbuf (37*512 halves). total ~10.4 MB

// WMMA-fragment weight buffer: 37 tiles of 512 halves.
// forward tile idx = FW_BASE[l] + nt*FW_KTN[l] + kt ; backward likewise.
__device__ __host__ __forceinline__ int fw_base(int l){ const int t[4]={0,12,16,17}; return t[l]; }
__device__ __host__ __forceinline__ int fw_ktn (int l){ const int t[4]={3,2,1,1};   return t[l]; }
__device__ __host__ __forceinline__ int bw_base(int l){ const int t[4]={25,21,19,18};return t[l]; }
__device__ __host__ __forceinline__ int bw_ktn (int l){ const int t[4]={2,1,1,1};   return t[l]; }

// ================= WMMA fragment helpers (wave32, 16x16x32 f16) =========
__device__ __forceinline__ int frag_k(int e) {
  int v = e >> 1;
  return ((v & 4) ? (2*v + 8) : (2*v)) + (e & 1);
}

// Fast A fragment: per CDNA5 layout each lane's 16 halves are two contiguous
// 8-half (16-byte) runs at k = kb+[0,8) and kb+16+[0,8)  ->  2 x ds_load_b128.
__device__ __forceinline__ v16h load_a_lds(const _Float16* X, int Kin, int ko, int lane) {
  int m  = lane & 15;
  int kb = (lane & 16) ? 8 : 0;
  const _Float16* row = X + m*Kin + ko + kb;
  v16h a;
  if (ko + kb + 8 <= Kin) {
    v8h lo = *(const v8h*)row;
#pragma unroll
    for (int e = 0; e < 8; ++e) a[e] = lo[e];
  } else {
#pragma unroll
    for (int e = 0; e < 8; ++e) a[e] = (_Float16)0.0f;
  }
  if (ko + kb + 24 <= Kin) {
    v8h hi = *(const v8h*)(row + 16);
#pragma unroll
    for (int e = 0; e < 8; ++e) a[8+e] = hi[e];
  } else {
#pragma unroll
    for (int e = 0; e < 8; ++e) a[8+e] = (_Float16)0.0f;
  }
  return a;
}

// Pre-swizzled B fragment: one contiguous 32-byte load per lane.
__device__ __forceinline__ v16h load_b_frag(const _Float16* __restrict__ wbuf,
                                            int tile, int lane) {
  return *(const v16h*)(wbuf + tile*512 + lane*16);
}

// gather-style builders (used only by the one-time prep kernel)
__device__ __forceinline__ v16h build_b_w(const float* W, int Kin, int Nout,
                                          int ko, int no, int lane) {
  int n  = no + (lane & 15);
  int kb = (lane & 16) ? 8 : 0;
  v16h b;
#pragma unroll
  for (int e = 0; e < 16; ++e) {
    int k = ko + kb + frag_k(e);
    float w = (k < Kin && n < Nout) ? W[k*Nout + n] : 0.0f;
    b[e] = (_Float16)w;
  }
  return b;
}
__device__ __forceinline__ v16h build_b_wT(const float* W, int din, int Kd,
                                           int ko, int no, int lane) {
  int n  = no + (lane & 15);
  int kb = (lane & 16) ? 8 : 0;
  v16h b;
#pragma unroll
  for (int e = 0; e < 16; ++e) {
    int k = ko + kb + frag_k(e);
    float w = (k < Kd && n < din) ? W[n*Kd + k] : 0.0f;
    b[e] = (_Float16)w;
  }
  return b;
}

__device__ __forceinline__ v8f wmma16(v16h a, v16h b, v8f c) {
  return __builtin_amdgcn_wmma_f32_16x16x32_f16(false, a, false, b, (short)0, c,
                                                false, false);
}

// Forward fitting layer: H = tanh(X @ W + b)
__device__ __forceinline__ void fit_fwd_layer(const _Float16* Xl, int Kin,
                                              const _Float16* __restrict__ wbuf, int l,
                                              const float* bias,
                                              _Float16* Hout, int Nout, int lane) {
  int ktn = fw_ktn(l);
  int ntn = (Nout + 15) / 16;
  for (int nt = 0; nt < ntn; ++nt) {
    v8f acc = {0.f,0.f,0.f,0.f,0.f,0.f,0.f,0.f};
    for (int kt = 0; kt < ktn; ++kt) {
      v16h a = load_a_lds(Xl, Kin, kt*32, lane);
      v16h b = load_b_frag(wbuf, fw_base(l) + nt*ktn + kt, lane);
      acc = wmma16(a, b, acc);
    }
    int n = (lane & 15) + nt*16;
    int mbase = (lane & 16) ? 8 : 0;
    if (n < Nout) {
      float bv = bias[n];
#pragma unroll
      for (int r = 0; r < 8; ++r) {
        float y = tanhf(acc[r] + bv);
        Hout[(mbase + r)*Nout + n] = (_Float16)y;
      }
    }
  }
}

// Backward fitting layer: Gout = (Gl @ W^T) * (1 - Hprev^2)
__device__ __forceinline__ void fit_bwd_layer(const _Float16* Gl, int Kd,
                                              const _Float16* __restrict__ wbuf, int l,
                                              int din, const _Float16* Hprev,
                                              _Float16* Gout, int lane) {
  int ktn = bw_ktn(l);
  int ntn = (din + 15) / 16;
  for (int nt = 0; nt < ntn; ++nt) {
    v8f acc = {0.f,0.f,0.f,0.f,0.f,0.f,0.f,0.f};
    for (int kt = 0; kt < ktn; ++kt) {
      v16h a = load_a_lds(Gl, Kd, kt*32, lane);
      v16h b = load_b_frag(wbuf, bw_base(l) + nt*ktn + kt, lane);
      acc = wmma16(a, b, acc);
    }
    int n = (lane & 15) + nt*16;
    int mbase = (lane & 16) ? 8 : 0;
    if (n < din) {
#pragma unroll
      for (int r = 0; r < 8; ++r) {
        int m = mbase + r;
        float h = (float)Hprev[m*din + n];
        Gout[m*din + n] = (_Float16)(acc[r] * (1.0f - h*h));
      }
    }
  }
}

// ================= scalar pyramid nets (1-/2-input, widths 2,4,8,16,32) ====
__device__ void net5_fwd1_tanh(float x, const Net5& n, float out[32]) {
  const int dims[5] = {2,4,8,16,32};
  float a[32], c[32];
  a[0] = x;
  int din = 1;
  for (int l = 0; l < 5; ++l) {
    int dout = dims[l];
    const float* W = n.W[l]; const float* b = n.b[l];
    for (int j = 0; j < dout; ++j) {
      float s = b[j];
      for (int i = 0; i < din; ++i) s += a[i] * W[i*dout + j];
      c[j] = tanhf(s);
    }
    for (int j = 0; j < dout; ++j) a[j] = c[j];
    din = dout;
  }
  for (int j = 0; j < 32; ++j) out[j] = a[j];
}

__device__ void net5_jvp1_tanh(float x, const Net5& n, float out[32], float tout[32]) {
  const int dims[5] = {2,4,8,16,32};
  float a[32], ta[32], c[32], tc[32];
  a[0] = x; ta[0] = 1.0f;
  int din = 1;
  for (int l = 0; l < 5; ++l) {
    int dout = dims[l];
    const float* W = n.W[l]; const float* b = n.b[l];
    for (int j = 0; j < dout; ++j) {
      float s = b[j], ts = 0.0f;
      for (int i = 0; i < din; ++i) {
        float w = W[i*dout + j];
        s  += a[i]  * w;
        ts += ta[i] * w;
      }
      float h = tanhf(s);
      c[j]  = h;
      tc[j] = (1.0f - h*h) * ts;
    }
    for (int j = 0; j < dout; ++j) { a[j] = c[j]; ta[j] = tc[j]; }
    din = dout;
  }
  for (int j = 0; j < 32; ++j) { out[j] = a[j]; tout[j] = ta[j]; }
}

__device__ void net5_fwd2_relu(float x0, float x1, const Net5& n, float out[32]) {
  const int dims[5] = {2,4,8,16,32};
  float a[32], c[32];
  a[0] = x0; a[1] = x1;
  int din = 2;
  for (int l = 0; l < 5; ++l) {
    int dout = dims[l];
    const float* W = n.W[l]; const float* b = n.b[l];
    for (int j = 0; j < dout; ++j) {
      float s = b[j];
      for (int i = 0; i < din; ++i) s += a[i] * W[i*dout + j];
      c[j] = fmaxf(s, 0.0f);
    }
    for (int j = 0; j < dout; ++j) a[j] = c[j];
    din = dout;
  }
  for (int j = 0; j < 32; ++j) out[j] = a[j];
}

__device__ void net5_fwd2_relu_all(float x0, float x1, const Net5& n, float hs[5][32]) {
  const int dims[5] = {2,4,8,16,32};
  float a[32];
  a[0] = x0; a[1] = x1;
  int din = 2;
  for (int l = 0; l < 5; ++l) {
    int dout = dims[l];
    const float* W = n.W[l]; const float* b = n.b[l];
    for (int j = 0; j < dout; ++j) {
      float s = b[j];
      for (int i = 0; i < din; ++i) s += a[i] * W[i*dout + j];
      hs[l][j] = fmaxf(s, 0.0f);
    }
    for (int j = 0; j < dout; ++j) a[j] = hs[l][j];
    din = dout;
  }
}

// ======================= kernels =======================

// K0: one-time pre-swizzle of fitting weights into WMMA fragment order (f16).
__global__ __launch_bounds__(32)
void k_prep(FitNet f, _Float16* __restrict__ wbuf) {
  int tile = blockIdx.x;    // 0..36
  int lane = threadIdx.x;
  const int KinA[4]  = {96,64,32,16};
  const int NoutA[4] = {64,32,16, 8};
  int l, kt, nt; bool bwd;
  if (tile < 12)      { bwd=false; l=0; int t=tile;    kt=t%3; nt=t/3; }
  else if (tile < 16) { bwd=false; l=1; int t=tile-12; kt=t%2; nt=t/2; }
  else if (tile < 17) { bwd=false; l=2; kt=0; nt=0; }
  else if (tile < 18) { bwd=false; l=3; kt=0; nt=0; }
  else if (tile < 19) { bwd=true;  l=3; kt=0; nt=0; }
  else if (tile < 21) { bwd=true;  l=2; kt=0; nt=tile-19; }
  else if (tile < 25) { bwd=true;  l=1; kt=0; nt=tile-21; }
  else                { bwd=true;  l=0; int t=tile-25; kt=t%2; nt=t/2; }
  v16h b = bwd ? build_b_wT(f.W[l], KinA[l], NoutA[l], kt*32, nt*16, lane)
               : build_b_w (f.W[l], KinA[l], NoutA[l], kt*32, nt*16, lane);
  *(v16h*)(wbuf + tile*512 + lane*16) = b;
}

// K1: type-1 NUFFT: fre[s,k] = sum_p cos(x_p k), fim[s,k] = -sum_p sin(x_p k)
__global__ void k_nufft_fwd(const float* __restrict__ R, float* fre, float* fim) {
  int s = blockIdx.x;
  __shared__ float xs[NPTS];
  for (int p = threadIdx.x; p < NPTS; p += blockDim.x) xs[p] = R[s*NPTS + p];
  __syncthreads();
  int k = blockIdx.y*blockDim.x + threadIdx.x;
  if (k >= NFK) return;
  float kv = KSTEP * (float)(k - NFK/2);
  float cr = 0.f, ci = 0.f;
  for (int p = 0; p < NPTS; ++p) {
    float sn, cs;
    __sincosf(xs[p]*kv, &sn, &cs);
    cr += cs; ci += sn;
  }
  fre[s*NFK + k] = cr;
  fim[s*NFK + k] = -ci;
}

// K2: type-2 NUFFT: u[s,p] = (1/NF) sum_k e^{+ikx_p} * (fmult * f_hat)
__global__ void k_nufft_type2(const float* __restrict__ R,
                              const float* __restrict__ fre, const float* __restrict__ fim,
                              const float* __restrict__ fmult,
                              float* ure, float* uim) {
  int s = blockIdx.x;
  __shared__ float gre[NFK], gim[NFK];
  for (int k = threadIdx.x; k < NFK; k += blockDim.x) {
    float m = fmult[k];
    gre[k] = m * fre[s*NFK + k];
    gim[k] = m * fim[s*NFK + k];
  }
  __syncthreads();
  int p = blockIdx.y*blockDim.x + threadIdx.x;
  if (p >= NPTS) return;
  float x = R[s*NPTS + p];
  float ar = 0.f, ai = 0.f;
  for (int k = 0; k < NFK; ++k) {
    float kv = KSTEP * (float)(k - NFK/2);
    float sn, cs;
    __sincosf(x*kv, &sn, &cs);
    ar += cs*gre[k] - sn*gim[k];
    ai += cs*gim[k] + sn*gre[k];
  }
  ure[s*NPTS + p] = ar * (1.0f/NFK);
  uim[s*NPTS + p] = ai * (1.0f/NFK);
}

// K3: per-atom forward: writes fused f16 fitting input row X0 = [D(64) | L3(32)]
__global__ void k_atom_fwd(const float* __restrict__ R, Net5 p1, Net5 p2, Net5 p3,
                           const float* __restrict__ ure, const float* __restrict__ uim,
                           _Float16* __restrict__ X0ws) {
  int idx = blockIdx.x*blockDim.x + threadIdx.x;
  if (idx >= NATOM) return;
  int s = idx / NPTS, p = idx % NPTS, c = p / NPP, i = p % NPP;
  float x = R[idx];
  float Dacc[64];
  for (int j = 0; j < 64; ++j) Dacc[j] = 0.f;
  float P[32];
  for (int t = 0; t < 3*NPP; ++t) {
    if (t == NPP + i) continue;
    int cc = c + t/NPP - 1, slot = t % NPP;
    float shift = 0.f;
    if (cc < 0)            { cc += NCELLS; shift = -LBOX; }
    else if (cc >= NCELLS) { cc -= NCELLS; shift =  LBOX; }
    float nb = R[s*NPTS + cc*NPP + slot] + shift;
    float r = fabsf(nb - x);
    float col0 = 1.0f / r;     // (1/r - AV0)/STD0
    float col1 = r;            // (r   - AV1)/STD1
    net5_fwd1_tanh(col1, p1, P);
    for (int j = 0; j < 32; ++j) Dacc[j] += P[j]*col1;
    net5_fwd1_tanh(col0, p2, P);
    for (int j = 0; j < 32; ++j) Dacc[32+j] += P[j]*col0;
  }
  _Float16* row = X0ws + (size_t)idx*96;
  for (int j = 0; j < 64; ++j) row[j] = (_Float16)Dacc[j];
  float lr0 = (ure[idx] - LRAV0)/LRSTD0;
  float lr1 = (uim[idx] - LRAV1)/LRSTD1;
  float h5[32];
  net5_fwd2_relu(lr0, lr1, p3, h5);
  for (int j = 0; j < 32; ++j) row[64+j] = (_Float16)h5[j];
}

// K4: fitting net 96->64->32->16->8 forward + full backward via WMMA.
// One wave per 16-atom tile; X0 tile staged with async global->LDS when available.
__global__ __launch_bounds__(32)
void k_fit(const _Float16* __restrict__ X0ws, const _Float16* __restrict__ wbuf,
           FitNet f, float* gD, float* gL3) {
  __shared__ __align__(16) _Float16 X0[16*96], H1[16*64], H2[16*32], H3[16*16], H4[16*8];
  __shared__ __align__(16) _Float16 Ga[16*96], Gb[16*96];
  int lane  = threadIdx.x;
  int atom0 = blockIdx.x * 16;
  const _Float16* gsrc = X0ws + (size_t)atom0*96;   // 3072 contiguous bytes
  __builtin_prefetch(wbuf, 0, 2);                   // global_prefetch_b8
#if USE_ASYNC_LDS
  {
#pragma unroll
    for (int it = 0; it < 6; ++it) {
      int off = (it*32 + lane) * 8;                 // 16-byte chunks
      __builtin_amdgcn_global_load_async_to_lds_b128(
          (__attribute__((address_space(1))) v4i*)(gsrc + off),
          (__attribute__((address_space(3))) v4i*)(X0 + off), 0, 0);
    }
    __builtin_amdgcn_s_wait_asynccnt(0);
  }
#else
  for (int t = lane; t < 16*96/8; t += 32)
    *(v8h*)(X0 + t*8) = *(const v8h*)(gsrc + t*8);
#endif
  __syncthreads();
  fit_fwd_layer(X0, 96, wbuf, 0, f.b[0], H1, 64, lane); __syncthreads();
  fit_fwd_layer(H1, 64, wbuf, 1, f.b[1], H2, 32, lane); __syncthreads();
  fit_fwd_layer(H2, 32, wbuf, 2, f.b[2], H3, 16, lane); __syncthreads();
  fit_fwd_layer(H3, 16, wbuf, 3, f.b[3], H4,  8, lane); __syncthreads();
  // seed: ghat4 = lw * (1 - h4^2)   (dE/dF == 1 per atom)
  for (int t = lane; t < 16*8; t += 32) {
    int m = t / 8, k = t % 8;
    float h = (float)H4[m*8 + k];
    Ga[m*8 + k] = (_Float16)(f.lw[k] * (1.0f - h*h));
  }
  __syncthreads();
  fit_bwd_layer(Ga,  8, wbuf, 3, 16, H3, Gb, lane); __syncthreads();
  fit_bwd_layer(Gb, 16, wbuf, 2, 32, H2, Ga, lane); __syncthreads();
  fit_bwd_layer(Ga, 32, wbuf, 1, 64, H1, Gb, lane); __syncthreads();
  // final: gX0 = ghat1 @ W1^T -> split into gD (0..63) and gL3 (64..95)
  for (int nt = 0; nt < 6; ++nt) {
    v8f acc = {0.f,0.f,0.f,0.f,0.f,0.f,0.f,0.f};
    for (int kt = 0; kt < 2; ++kt) {
      v16h a = load_a_lds(Gb, 64, kt*32, lane);
      v16h b = load_b_frag(wbuf, bw_base(0) + nt*2 + kt, lane);
      acc = wmma16(a, b, acc);
    }
    int n = (lane & 15) + nt*16;
    int mbase = (lane & 16) ? 8 : 0;
#pragma unroll
    for (int r = 0; r < 8; ++r) {
      int m = mbase + r;
      float g = acc[r];
      if (n < 64) gD[(atom0+m)*64 + n] = g;
      else        gL3[(atom0+m)*32 + (n-64)] = g;
    }
  }
}

// K5: backward through p3 (relu) + lr normalization -> gu_re, gu_im per atom
__global__ void k_p3_bwd(const float* __restrict__ ure, const float* __restrict__ uim,
                         Net5 p3, const float* __restrict__ gL3,
                         float* gur, float* gui) {
  int idx = blockIdx.x*blockDim.x + threadIdx.x;
  if (idx >= NATOM) return;
  const int dims[5] = {2,4,8,16,32};
  float lr0 = (ure[idx] - LRAV0)/LRSTD0;
  float lr1 = (uim[idx] - LRAV1)/LRSTD1;
  float hs[5][32];
  net5_fwd2_relu_all(lr0, lr1, p3, hs);
  float g[32], gz[32], gp[32];
  for (int j = 0; j < 32; ++j) g[j] = gL3[idx*32 + j];
  for (int l = 4; l >= 0; --l) {
    int dout = dims[l];
    int din  = (l == 0) ? 2 : dims[l-1];
    const float* W = p3.W[l];
    for (int j = 0; j < dout; ++j) gz[j] = (hs[l][j] > 0.0f) ? g[j] : 0.0f;
    for (int i = 0; i < din; ++i) {
      float s = 0.0f;
      for (int j = 0; j < dout; ++j) s += W[i*dout + j] * gz[j];
      gp[i] = s;
    }
    for (int i = 0; i < din; ++i) g[i] = gp[i];
  }
  gur[idx] = g[0] / LRSTD0;
  gui[idx] = g[1] / LRSTD1;
}

// K6: gu-weighted trig sums and per-(s,k) backward coefficients
__global__ void k_bwd_sums(const float* __restrict__ R,
                           const float* __restrict__ gur, const float* __restrict__ gui,
                           const float* __restrict__ fre, const float* __restrict__ fim,
                           const float* __restrict__ fmult,
                           float* coefC, float* coefS, float* mkA, float* mkB) {
  int s = blockIdx.x;
  __shared__ float xs[NPTS], wr[NPTS], wi[NPTS];
  for (int p = threadIdx.x; p < NPTS; p += blockDim.x) {
    xs[p] = R[s*NPTS + p];
    wr[p] = gur[s*NPTS + p];
    wi[p] = gui[s*NPTS + p];
  }
  __syncthreads();
  int k = blockIdx.y*blockDim.x + threadIdx.x;
  if (k >= NFK) return;
  float kv = KSTEP * (float)(k - NFK/2);
  float Ar=0.f, Br=0.f, Ai=0.f, Bi=0.f;
  for (int p = 0; p < NPTS; ++p) {
    float sn, cs;
    __sincosf(xs[p]*kv, &sn, &cs);
    Ar += wr[p]*cs; Br += wr[p]*sn;
    Ai += wi[p]*cs; Bi += wi[p]*sn;
  }
  float mk = fmult[k] * kv;
  float A = fre[s*NFK + k];
  float B = -fim[s*NFK + k];
  coefC[s*NFK + k] = mk * (Br - Ai);
  coefS[s*NFK + k] = mk * (-Ar - Bi);
  mkA[s*NFK + k]   = mk * A;
  mkB[s*NFK + k]   = mk * B;
}

// K7: long-range force per atom (writes every out element -> initializes d_out)
__global__ void k_lr_force(const float* __restrict__ R,
                           const float* __restrict__ gur, const float* __restrict__ gui,
                           const float* __restrict__ coefC, const float* __restrict__ coefS,
                           const float* __restrict__ mkA, const float* __restrict__ mkB,
                           float* out) {
  int s = blockIdx.x;
  __shared__ float cC[NFK], cS[NFK], mA[NFK], mB[NFK];
  for (int k = threadIdx.x; k < NFK; k += blockDim.x) {
    cC[k] = coefC[s*NFK + k];
    cS[k] = coefS[s*NFK + k];
    mA[k] = mkA[s*NFK + k];
    mB[k] = mkB[s*NFK + k];
  }
  __syncthreads();
  int p = blockIdx.y*blockDim.x + threadIdx.x;
  if (p >= NPTS) return;
  int idx = s*NPTS + p;
  float x = R[idx], gr = gur[idx], gi = gui[idx];
  float acc = 0.f;
  for (int k = 0; k < NFK; ++k) {
    float kv = KSTEP * (float)(k - NFK/2);
    float sn, cs;
    __sincosf(x*kv, &sn, &cs);
    acc += cs*( gr*mB[k] + gi*mA[k] + cC[k])
         + sn*(-gr*mA[k] + gi*mB[k] + cS[k]);
  }
  out[idx] = -(acc * (1.0f/NFK));   // force = -grad
}

// K8: short-range force: JVP through pyramids, scatter to self + neighbor
__global__ void k_sr_force(const float* __restrict__ R, Net5 p1, Net5 p2,
                           const float* __restrict__ gD, float* out) {
  int idx = blockIdx.x*blockDim.x + threadIdx.x;
  if (idx >= NATOM) return;
  int s = idx / NPTS, p = idx % NPTS, c = p / NPP, i = p % NPP;
  float x = R[idx];
  float gDl[64];
  for (int j = 0; j < 64; ++j) gDl[j] = gD[idx*64 + j];
  float P[32], T[32];
  float fq = 0.f;
  for (int t = 0; t < 3*NPP; ++t) {
    if (t == NPP + i) continue;
    int cc = c + t/NPP - 1, slot = t % NPP;
    float shift = 0.f;
    if (cc < 0)            { cc += NCELLS; shift = -LBOX; }
    else if (cc >= NCELLS) { cc -= NCELLS; shift =  LBOX; }
    int nbidx = s*NPTS + cc*NPP + slot;
    float nb = R[nbidx] + shift;
    float d = nb - x;
    float r = fabsf(d);
    float col0 = 1.0f / r;
    float col1 = r;
    net5_jvp1_tanh(col1, p1, P, T);
    float g1 = 0.f;
    for (int j = 0; j < 32; ++j) g1 += gDl[j] * (P[j] + col1*T[j]);
    net5_jvp1_tanh(col0, p2, P, T);
    float g0 = 0.f;
    for (int j = 0; j < 32; ++j) g0 += gDl[32+j] * (P[j] + col0*T[j]);
    // dE/dr = g1 * dcol1/dr + g0 * dcol0/dr ; dcol0/dr = -1/r^2
    float grd = g1 - g0/(r*r);
    float sgn = (d >= 0.f) ? 1.f : -1.f;
    fq += sgn * grd;                      // force on self
    atomicAdd(&out[nbidx], -sgn * grd);   // force on neighbor
  }
  atomicAdd(&out[idx], fq);
}

// ======================= launcher =======================
extern "C" void kernel_launch(void* const* d_in, const int* in_sizes, int n_in,
                              void* d_out, int out_size, void* d_ws, size_t ws_size,
                              hipStream_t stream) {
  const float* R = (const float*)d_in[0];
  Net5 p1, p2, p3;
  for (int l = 0; l < 5; ++l) {
    p1.W[l] = (const float*)d_in[1+l];  p1.b[l] = (const float*)d_in[6+l];
    p2.W[l] = (const float*)d_in[11+l]; p2.b[l] = (const float*)d_in[16+l];
    p3.W[l] = (const float*)d_in[21+l]; p3.b[l] = (const float*)d_in[26+l];
  }
  FitNet fit;
  for (int l = 0; l < 4; ++l) {
    fit.W[l] = (const float*)d_in[31+l];
    fit.b[l] = (const float*)d_in[35+l];
  }
  fit.lw = (const float*)d_in[39];
  const float* fmult = (const float*)d_in[41];

  float* ws  = (float*)d_ws;     // needs ~10.5 MB
  float* out = (float*)d_out;

  float* fre = ws + WS_FRE;  float* fim = ws + WS_FIM;
  float* ure = ws + WS_URE;  float* uim = ws + WS_UIM;
  float* gDw = ws + WS_GD;   float* gL3 = ws + WS_GL3;
  float* gur = ws + WS_GUR;  float* gui = ws + WS_GUI;
  float* cC  = ws + WS_CC;   float* cS  = ws + WS_CS;
  float* mkA = ws + WS_MKA;  float* mkB = ws + WS_MKB;
  _Float16* x0ws = (_Float16*)(ws + WS_F16);
  _Float16* wbuf = x0ws + (size_t)NATOM*96;

  dim3 gK(NSAMP, (NFK + 255)/256);
  dim3 gP(NSAMP, (NPTS + 255)/256);
  int atomBlocks = (NATOM + 127)/128;

  k_prep       <<<37, 32, 0, stream>>>(fit, wbuf);
  k_nufft_fwd  <<<gK, 256, 0, stream>>>(R, fre, fim);
  k_nufft_type2<<<gP, 256, 0, stream>>>(R, fre, fim, fmult, ure, uim);
  k_atom_fwd   <<<atomBlocks, 128, 0, stream>>>(R, p1, p2, p3, ure, uim, x0ws);
  k_fit        <<<NATOM/16, 32, 0, stream>>>(x0ws, wbuf, fit, gDw, gL3);
  k_p3_bwd     <<<atomBlocks, 128, 0, stream>>>(ure, uim, p3, gL3, gur, gui);
  k_bwd_sums   <<<gK, 256, 0, stream>>>(R, gur, gui, fre, fim, fmult, cC, cS, mkA, mkB);
  k_lr_force   <<<gP, 256, 0, stream>>>(R, gur, gui, cC, cS, mkA, mkB, out);
  k_sr_force   <<<atomBlocks, 128, 0, stream>>>(R, p1, p2, gDw, out);
}
